// SlotAttentionWithFeedbackIterQuant_58076547776907
// MI455X (gfx1250) — matrix operand order repaired and checked by reading, not compiled, not run on hardware
//
#include <hip/hip_runtime.h>

#define EPS_ 1e-8f
#define LNEPS_ 1e-5f
#define SCALE_ 0.125f   // DH^-0.5 = 64^-0.5

typedef __attribute__((ext_vector_type(16))) __bf16 v16bf;
typedef __attribute__((ext_vector_type(8)))  __bf16 v8bf;
typedef __attribute__((ext_vector_type(8)))  float  v8f;

static __device__ __forceinline__ float wave_red_add(float v) {
  v += __shfl_xor(v, 1, 32);
  v += __shfl_xor(v, 2, 32);
  v += __shfl_xor(v, 4, 32);
  v += __shfl_xor(v, 8, 32);
  v += __shfl_xor(v, 16, 32);
  return v;
}

// ---------------------------------------------------------------------------
// One-time convert [Wk;Wv] (fp32, 512x256) -> bf16 so the projection GEMM's
// B-fragments are pure b128 loads (no per-wave v_cvt in the hot loop).
// ---------------------------------------------------------------------------
__global__ void __launch_bounds__(256) k_cvt_w(
    const float* __restrict__ Wk, const float* __restrict__ Wv,
    __bf16* __restrict__ wbf) {
  const int row = blockIdx.x, t = threadIdx.x;
  const float* src = (row < 256) ? (Wk + (size_t)row * 256)
                                 : (Wv + (size_t)(row - 256) * 256);
  wbf[(size_t)row * 256 + t] = (__bf16)src[t];
}

// ---------------------------------------------------------------------------
// Fused LayerNorm(inputs) + K/V projection GEMM (bf16 WMMA, f32 accum).
// One wave owns a 16-row M-tile (full K=256 in-registers as 8 A-fragments),
// loops 32 N-tiles of the combined [Wk;Wv] weight (N=512).
// N-tile loop is NOT unrolled (keeps VGPRs < 256 -> real occupancy); each
// body half issues 8 b128 loads as a clause, then 4 WMMAs on 2 accumulators.
// K written row-major [row, 256]; V written TRANSPOSED vT[b][c][j].
// ---------------------------------------------------------------------------
__global__ void __launch_bounds__(256) k_ln_project(
    const float* __restrict__ inp, const float* __restrict__ g,
    const float* __restrict__ bta, const __bf16* __restrict__ wbf,
    __bf16* __restrict__ kbf, __bf16* __restrict__ vT) {
  __shared__ float sg[256], sb[256];
  sg[threadIdx.x] = g[threadIdx.x];
  sb[threadIdx.x] = bta[threadIdx.x];
  __syncthreads();
  const int wid = threadIdx.x >> 5, lane = threadIdx.x & 31;
  const int half = lane >> 4, m = lane & 15;
  const int mt = blockIdx.x * 8 + wid;          // 8192 M-tiles total
  const int row = mt * 16 + m;
  const float* rp = inp + (size_t)row * 256;

  v16bf a[8];
  float sum = 0.f, sq = 0.f;
#pragma unroll
  for (int kf = 0; kf < 8; ++kf) {
#pragma unroll
    for (int run = 0; run < 2; ++run) {
      const int col = kf * 32 + run * 16 + half * 8;
      const float4 f0 = *(const float4*)(rp + col);
      const float4 f1 = *(const float4*)(rp + col + 4);
      const float xs[8] = {f0.x, f0.y, f0.z, f0.w, f1.x, f1.y, f1.z, f1.w};
#pragma unroll
      for (int t = 0; t < 8; ++t) {
        sum += xs[t];
        sq  += xs[t] * xs[t];
        a[kf][run * 8 + t] = (__bf16)xs[t];
      }
    }
  }
  // a row is split between lane m and lane m+16 -> single xor-16 pairing
  sum += __shfl_xor(sum, 16, 32);
  sq  += __shfl_xor(sq, 16, 32);
  const float mean = sum * (1.f / 256.f);
  const float var  = sq * (1.f / 256.f) - mean * mean;
  const float rstd = rsqrtf(var + LNEPS_);
#pragma unroll
  for (int kf = 0; kf < 8; ++kf) {
#pragma unroll
    for (int run = 0; run < 2; ++run) {
      const int col = kf * 32 + run * 16 + half * 8;
#pragma unroll
      for (int t = 0; t < 8; ++t) {
        float f = (float)a[kf][run * 8 + t];
        f = (f - mean) * rstd * sg[col + t] + sb[col + t];
        a[kf][run * 8 + t] = (__bf16)f;
      }
    }
  }

  const int bb_ = (mt * 16) >> 12;   // batch index (M-tiles never straddle b)
#pragma unroll 1
  for (int nt = 0; nt < 32; ++nt) {
    const int n = nt * 16 + m;  // this lane's output column (B lane = N)
    const __bf16* wrow = wbf + (size_t)n * 256 + half * 8;
    v8f c0, c1;
#pragma unroll
    for (int r = 0; r < 8; ++r) { c0[r] = 0.f; c1[r] = 0.f; }
#pragma unroll
    for (int hk = 0; hk < 2; ++hk) {
      v16bf bfr[4];
#pragma unroll
      for (int kf = 0; kf < 4; ++kf) {
        const int kk = hk * 4 + kf;
        const v8bf r0 = *(const v8bf*)(wrow + kk * 32);
        const v8bf r1 = *(const v8bf*)(wrow + kk * 32 + 16);
#pragma unroll
        for (int t = 0; t < 8; ++t) { bfr[kf][t] = r0[t]; bfr[kf][8 + t] = r1[t]; }
      }
      c0 = __builtin_amdgcn_wmma_f32_16x16x32_bf16(false, a[hk * 4 + 0], false,
                                                   bfr[0], (short)0, c0, false,
                                                   false);
      c1 = __builtin_amdgcn_wmma_f32_16x16x32_bf16(false, a[hk * 4 + 1], false,
                                                   bfr[1], (short)0, c1, false,
                                                   false);
      c0 = __builtin_amdgcn_wmma_f32_16x16x32_bf16(false, a[hk * 4 + 2], false,
                                                   bfr[2], (short)0, c0, false,
                                                   false);
      c1 = __builtin_amdgcn_wmma_f32_16x16x32_bf16(false, a[hk * 4 + 3], false,
                                                   bfr[3], (short)0, c1, false,
                                                   false);
    }
#pragma unroll
    for (int r = 0; r < 8; ++r) {
      const int grow = mt * 16 + r + half * 8;  // C/D: VGPR r -> M = r + 8*halfsel
      const __bf16 ov = (__bf16)(c0[r] + c1[r]);
      if (n < 256) {
        kbf[(size_t)grow * 256 + n] = ov;
      } else {
        const int j = grow & 4095;
        vT[((size_t)bb_ * 256 + (n - 256)) * 4096 + j] = ov;
      }
    }
  }
}

// ---------------------------------------------------------------------------
// Generic row LayerNorm (f32 -> f32), wave per 256-wide row.
// ---------------------------------------------------------------------------
__global__ void __launch_bounds__(256) k_ln_rows(
    const float* __restrict__ src, const float* __restrict__ g,
    const float* __restrict__ bta, float* __restrict__ dst, int nrows) {
  const int wid = threadIdx.x >> 5, lane = threadIdx.x & 31;
  const int row = blockIdx.x * 8 + wid;
  if (row >= nrows) return;
  const float* rp = src + (size_t)row * 256;
  const int e0 = lane * 8;
  const float4 f0 = *(const float4*)(rp + e0);
  const float4 f1 = *(const float4*)(rp + e0 + 4);
  const float xs[8] = {f0.x, f0.y, f0.z, f0.w, f1.x, f1.y, f1.z, f1.w};
  float sum = 0.f, sq = 0.f;
#pragma unroll
  for (int t = 0; t < 8; ++t) { sum += xs[t]; sq += xs[t] * xs[t]; }
  sum = wave_red_add(sum);
  sq  = wave_red_add(sq);
  const float mean = sum * (1.f / 256.f);
  const float rstd = rsqrtf(sq * (1.f / 256.f) - mean * mean + LNEPS_);
  float* op = dst + (size_t)row * 256 + e0;
#pragma unroll
  for (int t = 0; t < 8; ++t)
    op[t] = (xs[t] - mean) * rstd * g[e0 + t] + bta[e0 + t];
}

// ---------------------------------------------------------------------------
// q = s @ Wq^T  -> bf16  (512 x 256, tiny; one block per row)
// ---------------------------------------------------------------------------
__global__ void __launch_bounds__(256) k_qproj(
    const float* __restrict__ s, const float* __restrict__ Wq,
    __bf16* __restrict__ qbf) {
  __shared__ float srow[256];
  const int row = blockIdx.x;
  srow[threadIdx.x] = s[(size_t)row * 256 + threadIdx.x];
  __syncthreads();
  const float* w = Wq + (size_t)threadIdx.x * 256;
  float acc = 0.f;
#pragma unroll 8
  for (int k = 0; k < 256; k += 4) {
    const float4 wv = *(const float4*)(w + k);
    acc += wv.x * srow[k] + wv.y * srow[k + 1] + wv.z * srow[k + 2] +
           wv.w * srow[k + 3];
  }
  qbf[(size_t)row * 256 + threadIdx.x] = (__bf16)acc;
}

// ---------------------------------------------------------------------------
// dots[b, i*4+h, j] = SCALE * sum_d q[b,i,h,d] * k[b,j,h,d]
// 1024 blocks (8192 waves ~ 8/SIMD); each wave does 4 consecutive j-tiles.
// k memory is [j,d] row-major == B^T, so B-fragments are contiguous per lane.
// ---------------------------------------------------------------------------
__global__ void __launch_bounds__(256) k_dots(
    const __bf16* __restrict__ qbf, const __bf16* __restrict__ kbf,
    float* __restrict__ dots) {
  const int b = blockIdx.x >> 5, h = (blockIdx.x >> 3) & 3;
  const int wid = threadIdx.x >> 5, lane = threadIdx.x & 31;
  const int half = lane >> 4, m = lane & 15;
  const int widx = (blockIdx.x & 7) * 8 + wid;   // 0..63 within (b,h)

  v16bf aq[2];
#pragma unroll
  for (int kf = 0; kf < 2; ++kf) {
    const __bf16* qp =
        qbf + (size_t)(b * 16 + m) * 256 + h * 64 + kf * 32 + half * 8;
    const v8bf r0 = *(const v8bf*)qp;
    const v8bf r1 = *(const v8bf*)(qp + 16);
#pragma unroll
    for (int t = 0; t < 8; ++t) { aq[kf][t] = r0[t]; aq[kf][8 + t] = r1[t]; }
  }

  v16bf bfr[4][2];
#pragma unroll
  for (int u = 0; u < 4; ++u) {
    const int j0 = (widx * 4 + u) * 16;
#pragma unroll
    for (int kf = 0; kf < 2; ++kf) {
      const __bf16* kp = kbf + (size_t)(b * 4096 + j0 + m) * 256 + h * 64 +
                         kf * 32 + half * 8;
      const v8bf r0 = *(const v8bf*)kp;
      const v8bf r1 = *(const v8bf*)(kp + 16);
#pragma unroll
      for (int t = 0; t < 8; ++t) { bfr[u][kf][t] = r0[t]; bfr[u][kf][8 + t] = r1[t]; }
    }
  }
  v8f c[4];
#pragma unroll
  for (int u = 0; u < 4; ++u) {
#pragma unroll
    for (int r = 0; r < 8; ++r) c[u][r] = 0.f;
    c[u] = __builtin_amdgcn_wmma_f32_16x16x32_bf16(false, aq[0], false,
                                                   bfr[u][0], (short)0, c[u],
                                                   false, false);
    c[u] = __builtin_amdgcn_wmma_f32_16x16x32_bf16(false, aq[1], false,
                                                   bfr[u][1], (short)0, c[u],
                                                   false, false);
  }
#pragma unroll
  for (int u = 0; u < 4; ++u) {
    const int j0 = (widx * 4 + u) * 16;
#pragma unroll
    for (int r = 0; r < 8; ++r) {
      const int i = r + half * 8;
      dots[((size_t)b * 64 + i * 4 + h) * 4096 + j0 + m] = c[u][r] * SCALE_;
    }
  }
}

// ---------------------------------------------------------------------------
// Column softmax over the 64 (= ns*H) axis per (b,j); in-place over dots.
// Per-block partial sums written to sumpart (no FP atomics -> deterministic).
// Last iteration also writes the head-mean attention output.
// ---------------------------------------------------------------------------
__global__ void __launch_bounds__(256) k_softmax(
    float* __restrict__ dots, float* __restrict__ sumpart,
    float* __restrict__ attn_out, int last) {
  const int idx = blockIdx.x * 256 + threadIdx.x;
  const int b = idx >> 12, j = idx & 4095;
  float x[64];
  float mx = -3.4e38f;
#pragma unroll
  for (int r = 0; r < 64; ++r) {
    x[r] = dots[((size_t)b * 64 + r) * 4096 + j];
    mx = fmaxf(mx, x[r]);
  }
  float s = 0.f;
#pragma unroll
  for (int r = 0; r < 64; ++r) { x[r] = __expf(x[r] - mx); s += x[r]; }
  const float rs = 1.f / s;

  __shared__ float sacc[8][64];
  const int lane = threadIdx.x & 31, wid = threadIdx.x >> 5;
#pragma unroll
  for (int r = 0; r < 64; ++r) {
    x[r] *= rs;
    dots[((size_t)b * 64 + r) * 4096 + j] = x[r];
    const float wsum = wave_red_add(x[r]);
    if (lane == 0) sacc[wid][r] = wsum;
  }
  if (last) {
#pragma unroll
    for (int i = 0; i < 16; ++i)
      attn_out[((size_t)b * 16 + i) * 4096 + j] =
          0.25f * (x[i * 4] + x[i * 4 + 1] + x[i * 4 + 2] + x[i * 4 + 3]);
  }
  __syncthreads();
  if (threadIdx.x < 64) {
    float acc = 0.f;
#pragma unroll
    for (int w = 0; w < 8; ++w) acc += sacc[w][threadIdx.x];
    sumpart[((b * 64 + threadIdx.x) << 4) + (blockIdx.x & 15)] = acc;
  }
}

__global__ void k_rdenom(const float* __restrict__ sumpart,
                         float* __restrict__ rden) {
  const int i = blockIdx.x * 256 + threadIdx.x;
  if (i < 2048) {
    const float* p = sumpart + ((size_t)i << 4);
    float s = 0.f;
#pragma unroll
    for (int t = 0; t < 16; ++t) s += p[t];
    rden[i] = 1.f / (s + 4096.f * EPS_);
  }
}

// ---------------------------------------------------------------------------
// updates partials: 8-way K-split over j (4096 waves ~ 4/SIMD), each wave
// covers 512 of 4096 j, inner loop unrolled x2 with two accumulators to
// break the serial WMMA accumulation chain. Partials summed in k_gru.
// ---------------------------------------------------------------------------
__global__ void __launch_bounds__(256) k_updates(
    const float* __restrict__ attn, const float* __restrict__ rden,
    const __bf16* __restrict__ vT, float* __restrict__ updp) {
  const int wid = threadIdx.x >> 5, lane = threadIdx.x & 31;
  const int w = blockIdx.x * 8 + wid;          // 4096 waves
  const int ks = w & 7, dt = (w >> 3) & 3, h = (w >> 5) & 3, b = w >> 7;
  const int half = lane >> 4, m = lane & 15;
  const int rowid = m * 4 + h;                 // attn row for slot i=m
  const float rd = rden[b * 64 + rowid];
  const float* ap = attn + ((size_t)b * 64 + rowid) * 4096;
  const int dcol = dt * 16 + m;                // B lane column (d within head)
  const __bf16* vp =
      vT + ((size_t)b * 256 + h * 64 + dcol) * 4096 + half * 8;

  v8f c0, c1;
#pragma unroll
  for (int r = 0; r < 8; ++r) { c0[r] = 0.f; c1[r] = 0.f; }

  const int jbase = ks * 512;
  for (int j0 = jbase; j0 < jbase + 512; j0 += 64) {
    __builtin_prefetch(vp + j0 + 64, 0, 3);
    // ---- sub-step 0 (j0) ----
    const float4 a0 = *(const float4*)(ap + j0 + half * 8);
    const float4 a1 = *(const float4*)(ap + j0 + half * 8 + 4);
    const float4 a2 = *(const float4*)(ap + j0 + 16 + half * 8);
    const float4 a3 = *(const float4*)(ap + j0 + 16 + half * 8 + 4);
    // ---- sub-step 1 (j0+32) ----
    const float4 b0 = *(const float4*)(ap + j0 + 32 + half * 8);
    const float4 b1 = *(const float4*)(ap + j0 + 32 + half * 8 + 4);
    const float4 b2 = *(const float4*)(ap + j0 + 48 + half * 8);
    const float4 b3 = *(const float4*)(ap + j0 + 48 + half * 8 + 4);
    const v8bf p00 = *(const v8bf*)(vp + j0);
    const v8bf p01 = *(const v8bf*)(vp + j0 + 16);
    const v8bf p10 = *(const v8bf*)(vp + j0 + 32);
    const v8bf p11 = *(const v8bf*)(vp + j0 + 48);

    const float av0[16] = {a0.x, a0.y, a0.z, a0.w, a1.x, a1.y, a1.z, a1.w,
                           a2.x, a2.y, a2.z, a2.w, a3.x, a3.y, a3.z, a3.w};
    const float av1[16] = {b0.x, b0.y, b0.z, b0.w, b1.x, b1.y, b1.z, b1.w,
                           b2.x, b2.y, b2.z, b2.w, b3.x, b3.y, b3.z, b3.w};
    v16bf aa0, aa1, bb0, bb1;
#pragma unroll
    for (int t = 0; t < 16; ++t) {
      aa0[t] = (__bf16)((av0[t] + EPS_) * rd);
      aa1[t] = (__bf16)((av1[t] + EPS_) * rd);
    }
#pragma unroll
    for (int t = 0; t < 8; ++t) {
      bb0[t] = p00[t]; bb0[8 + t] = p01[t];
      bb1[t] = p10[t]; bb1[8 + t] = p11[t];
    }
    c0 = __builtin_amdgcn_wmma_f32_16x16x32_bf16(false, aa0, false, bb0,
                                                 (short)0, c0, false, false);
    c1 = __builtin_amdgcn_wmma_f32_16x16x32_bf16(false, aa1, false, bb1,
                                                 (short)0, c1, false, false);
  }
#pragma unroll
  for (int r = 0; r < 8; ++r) {
    const int i = r + half * 8;
    updp[(size_t)ks * 131072 + ((size_t)b * 16 + i) * 256 + h * 64 + dt * 16 +
         m] = c0[r] + c1[r];
  }
}

// ---------------------------------------------------------------------------
// GRU cell (block per slot row); sums the 8 K-split update partials in fixed
// order during LDS staging. In-place safe w.r.t. hprev==out.
// ---------------------------------------------------------------------------
__global__ void __launch_bounds__(256) k_gru(
    const float* __restrict__ updp, const float* __restrict__ hprev,
    const float* __restrict__ Wih, const float* __restrict__ Whh,
    const float* __restrict__ bih, const float* __restrict__ bhh,
    float* __restrict__ out) {
  __shared__ float su[256], sh[256];
  const int row = blockIdx.x, j = threadIdx.x;
  const size_t o = (size_t)row * 256 + j;
  float us = 0.f;
#pragma unroll
  for (int k = 0; k < 8; ++k) us += updp[(size_t)k * 131072 + o];
  su[j] = us;
  sh[j] = hprev[o];
  __syncthreads();
  float gi[3], gh[3];
#pragma unroll
  for (int gate = 0; gate < 3; ++gate) {
    const float* wi = Wih + ((size_t)gate * 256 + j) * 256;
    const float* wh = Whh + ((size_t)gate * 256 + j) * 256;
    float ai = 0.f, ah = 0.f;
#pragma unroll 4
    for (int k = 0; k < 256; k += 4) {
      const float4 wiv = *(const float4*)(wi + k);
      const float4 whv = *(const float4*)(wh + k);
      ai += wiv.x * su[k] + wiv.y * su[k + 1] + wiv.z * su[k + 2] +
            wiv.w * su[k + 3];
      ah += whv.x * sh[k] + whv.y * sh[k + 1] + whv.z * sh[k + 2] +
            whv.w * sh[k + 3];
    }
    gi[gate] = ai + bih[gate * 256 + j];
    gh[gate] = ah + bhh[gate * 256 + j];
  }
  const float r = 1.f / (1.f + __expf(-(gi[0] + gh[0])));
  const float z = 1.f / (1.f + __expf(-(gi[1] + gh[1])));
  const float nn = tanhf(gi[2] + r * gh[2]);
  out[o] = (1.f - z) * nn + z * sh[j];
}

// ---------------------------------------------------------------------------
// VQ: argmin over 512 codes, quantize slots in place; per-row commit partial
// written to cpart (deterministic; summed in k_commit_final). Last iter also
// writes slots/inds outputs.
// ---------------------------------------------------------------------------
__global__ void __launch_bounds__(256) k_vq(
    float* __restrict__ slots, const float* __restrict__ cb, int last,
    float* __restrict__ out_slots, float* __restrict__ out_inds,
    float* __restrict__ cpart) {
  __shared__ float sx[256];
  __shared__ float sdist[256];
  __shared__ int   sidx[256];
  const int row = blockIdx.x, t = threadIdx.x;
  sx[t] = slots[(size_t)row * 256 + t];
  __syncthreads();
  float best = 3.4e38f;
  int bi = 0;
#pragma unroll
  for (int cc = 0; cc < 2; ++cc) {
    const int code = t + cc * 256;
    const float* cp = cb + (size_t)code * 256;
    float d2 = 0.f;
#pragma unroll 4
    for (int k = 0; k < 256; k += 4) {
      const float4 cv = *(const float4*)(cp + k);
      float dx = cv.x - sx[k];     d2 += dx * dx;
      dx = cv.y - sx[k + 1];       d2 += dx * dx;
      dx = cv.z - sx[k + 2];       d2 += dx * dx;
      dx = cv.w - sx[k + 3];       d2 += dx * dx;
    }
    if (d2 < best || (d2 == best && code < bi)) { best = d2; bi = code; }
  }
  sdist[t] = best;
  sidx[t] = bi;
  __syncthreads();
  for (int off = 128; off > 0; off >>= 1) {
    if (t < off) {
      const float od = sdist[t + off];
      const int oi = sidx[t + off];
      if (od < sdist[t] || (od == sdist[t] && oi < sidx[t])) {
        sdist[t] = od;
        sidx[t] = oi;
      }
    }
    __syncthreads();
  }
  const int ind = sidx[0];
  __syncthreads();
  const float qv = cb[(size_t)ind * 256 + t];
  slots[(size_t)row * 256 + t] = qv;
  const float diff = qv - sx[t];
  sdist[t] = diff * diff;
  __syncthreads();
  for (int off = 128; off > 0; off >>= 1) {
    if (t < off) sdist[t] += sdist[t + off];
    __syncthreads();
  }
  if (t == 0) cpart[row] = sdist[0];
  if (last) {
    out_slots[(size_t)row * 256 + t] = qv;
    if (t == 0) out_inds[row] = (float)ind;
  }
}

__global__ void k_commit_final(const float* __restrict__ cpart,
                               float* __restrict__ out) {
  float s = 0.f;
  for (int r = 0; r < 512; ++r) s += cpart[r];
  out[0] = s * (1.f / (512.f * 256.f));
}

// ---------------------------------------------------------------------------
extern "C" void kernel_launch(void* const* d_in, const int* in_sizes, int n_in,
                              void* d_out, int out_size, void* d_ws,
                              size_t ws_size, hipStream_t stream) {
  const float* inputs = (const float*)d_in[0];
  const float* cond   = (const float*)d_in[1];
  const float* g_in   = (const float*)d_in[2];
  const float* b_in   = (const float*)d_in[3];
  const float* g_slot = (const float*)d_in[4];
  const float* b_slot = (const float*)d_in[5];
  const float* Wq     = (const float*)d_in[6];
  const float* Wk     = (const float*)d_in[7];
  const float* Wv     = (const float*)d_in[8];
  const float* Wih    = (const float*)d_in[9];
  const float* Whh    = (const float*)d_in[10];
  const float* bih    = (const float*)d_in[11];
  const float* bhh    = (const float*)d_in[12];
  const float* cb     = (const float*)d_in[13];

  float* out = (float*)d_out;
  float* out_slots  = out;                         // 32*16*256
  float* out_attn   = out + 131072;                // 32*16*4096
  float* out_inds   = out_attn + 2097152;          // 512
  float* out_commit = out_inds + 512;              // 1

  char* ws = (char*)d_ws;
  __bf16* kbf  = (__bf16*)(ws);                    // 67108864 B
  __bf16* vT   = (__bf16*)(ws + 67108864);         // 67108864 B (transposed V)
  float*  dots = (float*)(ws + 134217728);         // 33554432 B (attn in place)
  float*  sump = (float*)(ws + 167772160);         // 131072 B (softmax partials)
  float*  rden = (float*)(ws + 167903232);         // 8 KB
  float*  sbuf = (float*)(ws + 167911424);         // 524288 B
  float*  slots= (float*)(ws + 168435712);         // 524288 B
  __bf16* qbf  = (__bf16*)(ws + 168960000);        // 262144 B
  float*  updp = (float*)(ws + 169222144);         // 4194304 B (8 K-split partials)
  float*  cprt = (float*)(ws + 173416448);         // 2048 B
  __bf16* wbf  = (__bf16*)(ws + 173418496);        // 262144 B (bf16 [Wk;Wv])

  k_cvt_w<<<512, 256, 0, stream>>>(Wk, Wv, wbf);
  k_ln_project<<<1024, 256, 0, stream>>>(inputs, g_in, b_in, wbf, kbf, vT);

  const float* hsrc = cond;
  for (int it = 0; it < 3; ++it) {
    const int last = (it == 2);
    k_ln_rows<<<64, 256, 0, stream>>>(hsrc, g_slot, b_slot, sbuf, 512);
    k_qproj<<<512, 256, 0, stream>>>(sbuf, Wq, qbf);
    k_dots<<<1024, 256, 0, stream>>>(qbf, kbf, dots);
    k_softmax<<<512, 256, 0, stream>>>(dots, sump, out_attn, last);
    k_rdenom<<<8, 256, 0, stream>>>(sump, rden);
    k_updates<<<512, 256, 0, stream>>>(dots, rden, vT, updp);
    k_gru<<<512, 256, 0, stream>>>(updp, hsrc, Wih, Whh, bih, bhh, slots);
    k_vq<<<512, 256, 0, stream>>>(slots, cb, last, out_slots, out_inds, cprt);
    hsrc = slots;
  }
  k_commit_final<<<1, 1, 0, stream>>>(cprt, out_commit);
}
